// MetacognitionModule_51049981281212
// MI455X (gfx1250) — matrix operand order repaired and checked by reading, not compiled, not run on hardware
//
#include <hip/hip_runtime.h>

// Problem constants (match the reference)
#define B_   8
#define S_   2048
#define H_   2048
#define M_   256
#define E_   8
#define HH_  1024   // H/2

// Workspace layout (requires ws_size >= ~128.1 MB):
//   [0, 32MB)          W1 packed bf16 B-fragments
//   [32MB, 64MB)       W2 packed bf16 B-fragments
//   [64MB, 128MB)      x packed bf16 A-fragments
//   [128MB, +64KB)     pooled [B,H] f32
//   then               wgt [B,E] f32
#define WS_W1B_OFF   ((size_t)0)
#define WS_W2B_OFF   ((size_t)32 * 1024 * 1024)
#define WS_XA_OFF    ((size_t)64 * 1024 * 1024)
#define WS_POOL_OFF  ((size_t)128 * 1024 * 1024)
#define WS_WGT_OFF   (WS_POOL_OFF + (size_t)64 * 1024)

#define MT_  32                      // tokens per block (two 16-row A tiles)
#define HEP_ (HH_ + 8)               // padded LDS row stride (bank-conflict-free)

typedef __attribute__((ext_vector_type(16))) __bf16 bf16x16;
typedef __attribute__((ext_vector_type(8)))  __bf16 bf16x8;
typedef __attribute__((ext_vector_type(8)))  float  f32x8;
typedef __attribute__((ext_vector_type(4)))  float  f32x4;

// ---------------------------------------------------------------------------
// Pack fp32 weight matrix W[e][K][N] (row-major, E experts) into bf16 WMMA
// B-fragments: layout [e][kt][ntile][lane(32)][16], where for the 32x16 bf16
// B-matrix, lane l holds column n = ntile*16 + (l&15), K values
// kt*32 + (l>>4)*8 + {0..7} and +16+{0..7}.  Each fragment is a fully
// coalesced 1KB wave read in the GEMM kernel.
// Grid: E * (Kdim/32) * (Ndim/128) blocks of 256 threads.
// ---------------------------------------------------------------------------
__global__ __launch_bounds__(256) void pack_weights_kernel(
    const float* __restrict__ W, __bf16* __restrict__ Wp, int Kdim, int Ndim)
{
    __shared__ __bf16 tile[32][136];   // 32K x 128N staging (+8 pad)
    const int ktiles  = Kdim >> 5;
    const int ngroups = Ndim >> 7;
    const int ng = blockIdx.x % ngroups;
    const int kt = (blockIdx.x / ngroups) % ktiles;
    const int e  =  blockIdx.x / (ngroups * ktiles);

    const float* src = W + ((size_t)e * Kdim + (size_t)kt * 32) * Ndim + (size_t)ng * 128;
    const int t  = threadIdx.x;
    const int r  = t >> 3;            // 0..31  (K row within tile)
    const int c0 = (t & 7) * 16;      // col start
#pragma unroll
    for (int j = 0; j < 16; ++j)
        tile[r][c0 + j] = (__bf16)src[(size_t)r * Ndim + c0 + j];
    __syncthreads();

    const int f    = t >> 5;          // fragment (ntile) within this 128-col group
    const int lane = t & 31;
    const int kloc = (lane >> 4) * 8;
    const int n    = f * 16 + (lane & 15);
    bf16x16 v;
#pragma unroll
    for (int j = 0; j < 16; ++j) v[j] = tile[(j < 8) ? (kloc + j) : (16 + kloc + j - 8)][n];

    const size_t base =
        ((((size_t)e * ktiles + kt) * (Ndim >> 4) + (size_t)ng * 8 + f) * 32 + lane) * 16;
    *(bf16x16*)(Wp + base) = v;
}

// ---------------------------------------------------------------------------
// Pack fp32 activations x[token][h] into bf16 WMMA A-fragments:
// layout [tokenTile(1024)][kt(64)][lane(32)][16].  For the 16x32 bf16 A
// matrix, lane l holds token row (l&15), K values kt*32 + (l>>4)*8 + {0..7}
// and +16+{0..7}.  One wave packs one fragment; writes are coalesced 1KB.
// Grid: (B*S/16) * (H/32) / 8 blocks of 256 threads (8 waves).
// ---------------------------------------------------------------------------
__global__ __launch_bounds__(256) void pack_x_kernel(
    const float* __restrict__ x, __bf16* __restrict__ xA)
{
    const int wid   = blockIdx.x * 8 + (threadIdx.x >> 5);  // global wave id
    const int lane  = threadIdx.x & 31;
    const int tile  = wid >> 6;            // token tile (H_/32 = 64 kt per tile)
    const int kt    = wid & 63;
    const int laneN = lane & 15;
    const int laneH = lane >> 4;

    const float* xp = x + (size_t)(tile * 16 + laneN) * H_ + kt * 32 + laneH * 8;
    const f32x4 x0 = *(const f32x4*)(xp);
    const f32x4 x1 = *(const f32x4*)(xp + 4);
    const f32x4 x2 = *(const f32x4*)(xp + 16);
    const f32x4 x3 = *(const f32x4*)(xp + 20);
    bf16x16 a;
#pragma unroll
    for (int j = 0; j < 4; ++j) {
        a[j]      = (__bf16)x0[j];
        a[4 + j]  = (__bf16)x1[j];
        a[8 + j]  = (__bf16)x2[j];
        a[12 + j] = (__bf16)x3[j];
    }
    *(bf16x16*)(xA + (((size_t)tile * 64 + kt) * 32 + lane) * 16) = a;
}

// ---------------------------------------------------------------------------
// pooled[b][h] = mean over s of x[b][s][h].   Grid: B*H/256 blocks.
// ---------------------------------------------------------------------------
__global__ __launch_bounds__(256) void pool_kernel(
    const float* __restrict__ x, float* __restrict__ pooled)
{
    const int idx = blockIdx.x * 256 + threadIdx.x;   // 0 .. B*H-1
    const int b = idx >> 11;                          // /H_ (=2048)
    const int h = idx & (H_ - 1);
    const float* p = x + (size_t)b * S_ * H_ + h;
    float s = 0.f;
    for (int ss = 0; ss < S_; ++ss) s += p[(size_t)ss * H_];
    pooled[idx] = s * (1.0f / (float)S_);
}

// ---------------------------------------------------------------------------
// Router: h1 = relu(pooled@Wm1+bm1); h2 = relu(h1@Wm2+bm2);
// probs = softmax(h2@Wm3+bm3);  w = softmax(probs*eff).  One block, 256 thr.
// ---------------------------------------------------------------------------
__global__ __launch_bounds__(256) void router_kernel(
    const float* __restrict__ pooled,
    const float* __restrict__ Wm1, const float* __restrict__ bm1,
    const float* __restrict__ Wm2, const float* __restrict__ bm2,
    const float* __restrict__ Wm3, const float* __restrict__ bm3,
    const float* __restrict__ eff, float* __restrict__ wgt)
{
    __shared__ float h1[B_][M_];
    __shared__ float h2[B_][M_];
    __shared__ float lg[B_][E_];
    const int t = threadIdx.x;   // t == output neuron m (0..255)

    {
        float s[B_];
#pragma unroll
        for (int b = 0; b < B_; ++b) s[b] = bm1[t];
        for (int h = 0; h < H_; ++h) {
            const float wv = Wm1[(size_t)h * M_ + t];
#pragma unroll
            for (int b = 0; b < B_; ++b) s[b] = fmaf(pooled[b * H_ + h], wv, s[b]);
        }
#pragma unroll
        for (int b = 0; b < B_; ++b) h1[b][t] = fmaxf(s[b], 0.f);
    }
    __syncthreads();
    {
        float s[B_];
#pragma unroll
        for (int b = 0; b < B_; ++b) s[b] = bm2[t];
        for (int m = 0; m < M_; ++m) {
            const float wv = Wm2[(size_t)m * M_ + t];
#pragma unroll
            for (int b = 0; b < B_; ++b) s[b] = fmaf(h1[b][m], wv, s[b]);
        }
#pragma unroll
        for (int b = 0; b < B_; ++b) h2[b][t] = fmaxf(s[b], 0.f);
    }
    __syncthreads();
    if (t < B_ * E_) {
        const int b = t >> 3, e = t & 7;
        float s = bm3[e];
        for (int m = 0; m < M_; ++m) s = fmaf(h2[b][m], Wm3[(size_t)m * E_ + e], s);
        lg[b][e] = s;
    }
    __syncthreads();
    if (t < B_) {
        const int b = t;
        float mx = -1e30f;
#pragma unroll
        for (int e = 0; e < E_; ++e) mx = fmaxf(mx, lg[b][e]);
        float p[E_]; float den = 0.f;
#pragma unroll
        for (int e = 0; e < E_; ++e) { p[e] = expf(lg[b][e] - mx); den += p[e]; }
        const float inv = 1.f / den;
#pragma unroll
        for (int e = 0; e < E_; ++e) p[e] = p[e] * inv * eff[e];
        mx = -1e30f;
#pragma unroll
        for (int e = 0; e < E_; ++e) mx = fmaxf(mx, p[e]);
        float q[E_]; float den2 = 0.f;
#pragma unroll
        for (int e = 0; e < E_; ++e) { q[e] = expf(p[e] - mx); den2 += q[e]; }
        const float inv2 = 1.f / den2;
#pragma unroll
        for (int e = 0; e < E_; ++e) wgt[b * E_ + e] = q[e] * inv2;
    }
}

// ---------------------------------------------------------------------------
// Fused expert MLP.  One block = 32 consecutive tokens (two 16-row A tiles),
// 8 waves (wave32).  Every B fragment loaded from L2 feeds TWO WMMAs (both
// M-tiles), halving weight-stream bandwidth demand vs. a 16-token tile.
// For each expert e:
//   GEMM1: he[32,1024] = relu(x_tile @ W1_e + b1_e)      (bf16 WMMA, f32 acc)
//          each wave owns 8 N-tiles (128 cols); he stored bf16 in LDS.
//   GEMM2: ye[32,2048] = tanh(he @ W2_e + b2_e)          (bf16 WMMA, f32 acc)
//          each wave owns 16 N-tiles (256 cols), done in 2 halves of 8.
//   out_tile += w[b,e] * ye   (block-private global RMW; e==0 initializes).
// Dynamic LDS: 32 * (1024+8) * 2 = 66048 bytes.
// ---------------------------------------------------------------------------
__global__ __launch_bounds__(256) void expert_moe_kernel(
    const __bf16* __restrict__ xA,
    const float*  __restrict__ b1,
    const float*  __restrict__ b2,
    const __bf16* __restrict__ W1B,
    const __bf16* __restrict__ W2B,
    const float*  __restrict__ wgt,
    float*        __restrict__ out)
{
    extern __shared__ __bf16 heS[];              // [MT_][HEP_] = 66 KB

    const int tile0  = blockIdx.x * 2;           // first 16-token tile
    const int token0 = blockIdx.x * MT_;
    const int b      = token0 >> 11;             // / S_
    const int wave   = threadIdx.x >> 5;
    const int lane   = threadIdx.x & 31;
    const int laneN  = lane & 15;
    const int laneH  = lane >> 4;

    for (int e = 0; e < E_; ++e) {
        const float we = wgt[b * E_ + e];

        // ---------------- GEMM1: x(32x2048) @ W1_e(2048x1024) ----------------
        f32x8 acc[2][8] = {};
        for (int kt = 0; kt < H_ / 32; ++kt) {
            const bf16x16 a0 = *(const bf16x16*)(
                xA + (((size_t)tile0 * 64 + kt) * 32 + lane) * 16);
            const bf16x16 a1 = *(const bf16x16*)(
                xA + ((((size_t)tile0 + 1) * 64 + kt) * 32 + lane) * 16);
            const __bf16* wp = W1B +
                ((((size_t)e * (H_ / 32) + kt) * (HH_ / 16) + (size_t)wave * 8) * 32 + lane) * 16;
#pragma unroll
            for (int nt = 0; nt < 8; ++nt) {
                const bf16x16 bf = *(const bf16x16*)(wp + (size_t)nt * 512);
                acc[0][nt] = __builtin_amdgcn_wmma_f32_16x16x32_bf16(
                    false, a0, false, bf, (short)0, acc[0][nt], false, false);
                acc[1][nt] = __builtin_amdgcn_wmma_f32_16x16x32_bf16(
                    false, a1, false, bf, (short)0, acc[1][nt], false, false);
            }
        }
        // epilogue: + b1, relu, -> bf16 LDS
#pragma unroll
        for (int nt = 0; nt < 8; ++nt) {
            const int n0 = (wave * 8 + nt) * 16;
            const float bias = b1[(size_t)e * HH_ + n0 + laneN];
#pragma unroll
            for (int m = 0; m < 2; ++m) {
#pragma unroll
                for (int r = 0; r < 8; ++r) {
                    const int row = m * 16 + r + laneH * 8;
                    const float v = acc[m][nt][r] + bias;
                    heS[row * HEP_ + n0 + laneN] = (__bf16)fmaxf(v, 0.f);
                }
            }
        }
        __syncthreads();

        // ---------------- GEMM2: he(32x1024) @ W2_e(1024x2048) ---------------
#pragma unroll 1
        for (int half = 0; half < 2; ++half) {
            f32x8 acc2[2][8] = {};
            for (int kt = 0; kt < HH_ / 32; ++kt) {
                bf16x16 a[2];
#pragma unroll
                for (int m = 0; m < 2; ++m) {
                    const __bf16* hp = heS + (m * 16 + laneN) * HEP_ + kt * 32 + laneH * 8;
                    const bf16x8 lo = *(const bf16x8*)(hp);
                    const bf16x8 hi = *(const bf16x8*)(hp + 16);
#pragma unroll
                    for (int j = 0; j < 8; ++j) { a[m][j] = lo[j]; a[m][8 + j] = hi[j]; }
                }
                const __bf16* wp = W2B +
                    ((((size_t)e * (HH_ / 32) + kt) * (H_ / 16)
                      + (size_t)wave * 16 + half * 8) * 32 + lane) * 16;
#pragma unroll
                for (int nt = 0; nt < 8; ++nt) {
                    const bf16x16 bf = *(const bf16x16*)(wp + (size_t)nt * 512);
                    acc2[0][nt] = __builtin_amdgcn_wmma_f32_16x16x32_bf16(
                        false, a[0], false, bf, (short)0, acc2[0][nt], false, false);
                    acc2[1][nt] = __builtin_amdgcn_wmma_f32_16x16x32_bf16(
                        false, a[1], false, bf, (short)0, acc2[1][nt], false, false);
                }
            }
            // epilogue: tanh(+b2), scale by w[b,e], accumulate block-private out tile
#pragma unroll
            for (int nt = 0; nt < 8; ++nt) {
                const int n0 = (wave * 16 + half * 8 + nt) * 16;
                const float bias = b2[(size_t)e * H_ + n0 + laneN];
#pragma unroll
                for (int m = 0; m < 2; ++m) {
#pragma unroll
                    for (int r = 0; r < 8; ++r) {
                        const int row = m * 16 + r + laneH * 8;
                        const size_t o = (size_t)(token0 + row) * H_ + n0 + laneN;
                        const float prev = (e == 0) ? 0.f : out[o];
                        out[o] = prev + we * tanhf(acc2[m][nt][r] + bias);
                    }
                }
            }
        }
        __syncthreads();   // heS reused by next expert
    }
}

// ---------------------------------------------------------------------------
extern "C" void kernel_launch(void* const* d_in, const int* in_sizes, int n_in,
                              void* d_out, int out_size, void* d_ws, size_t ws_size,
                              hipStream_t stream)
{
    const float* x   = (const float*)d_in[0];
    const float* Wm1 = (const float*)d_in[1];
    const float* bm1 = (const float*)d_in[2];
    const float* Wm2 = (const float*)d_in[3];
    const float* bm2 = (const float*)d_in[4];
    const float* Wm3 = (const float*)d_in[5];
    const float* bm3 = (const float*)d_in[6];
    const float* W1  = (const float*)d_in[7];
    const float* b1  = (const float*)d_in[8];
    const float* W2  = (const float*)d_in[9];
    const float* b2  = (const float*)d_in[10];
    const float* eff = (const float*)d_in[11];
    float* out = (float*)d_out;

    char* ws = (char*)d_ws;
    __bf16* W1B    = (__bf16*)(ws + WS_W1B_OFF);
    __bf16* W2B    = (__bf16*)(ws + WS_W2B_OFF);
    __bf16* xA     = (__bf16*)(ws + WS_XA_OFF);
    float*  pooled = (float*)(ws + WS_POOL_OFF);
    float*  wgt    = (float*)(ws + WS_WGT_OFF);

    // 1) Pack expert weights + activations into bf16 WMMA fragment layouts.
    pack_weights_kernel<<<E_ * (H_ / 32) * (HH_ / 128), 256, 0, stream>>>(W1, W1B, H_, HH_);
    pack_weights_kernel<<<E_ * (HH_ / 32) * (H_ / 128), 256, 0, stream>>>(W2, W2B, HH_, H_);
    pack_x_kernel<<<(B_ * S_ / 16) * (H_ / 32) / 8, 256, 0, stream>>>(x, xA);

    // 2) Router path.
    pool_kernel<<<(B_ * H_) / 256, 256, 0, stream>>>(x, pooled);
    router_kernel<<<1, 256, 0, stream>>>(pooled, Wm1, bm1, Wm2, bm2, Wm3, bm3, eff, wgt);

    // 3) Fused expert MLPs + weighted combine.
    const size_t ldsBytes = (size_t)MT_ * HEP_ * sizeof(__bf16);   // 66048 B
    expert_moe_kernel<<<(B_ * S_) / MT_, 256, ldsBytes, stream>>>(
        xA, b1, b2, W1B, W2B, wgt, out);
}
